// Graclus_26474178412616
// MI455X (gfx1250) — matrix operand order repaired and checked by reading, not compiled, not run on hardware
//
#include <hip/hip_runtime.h>

// ---------------------------------------------------------------------------
// Graclus + max_pool + pool_edge + GraphConv(relu) for MI455X (gfx1250).
// ---------------------------------------------------------------------------
#define HID 384          // hidden channels
#define DSLOT 64         // neighbor slots per node
#define MAXN_BITS 2048   // LDS matched-bitset words (covers N <= 65536)
#define LDSPAD 40        // halfs per LDS tile row (32 + 8 pad, 80B: 16B aligned)

typedef _Float16 half_t;
typedef __attribute__((ext_vector_type(16))) _Float16 v16h;
typedef __attribute__((ext_vector_type(8)))  _Float16 v8h;
typedef __attribute__((ext_vector_type(8)))  float    v8f;

#define INVALID_ID 0x7FFFFFFF

__device__ __forceinline__ unsigned enc_f32(float f) {
    unsigned u = __float_as_uint(f);
    return (u & 0x80000000u) ? ~u : (u | 0x80000000u);
}
__device__ __forceinline__ float dec_f32(unsigned u) {
    unsigned b = (u & 0x80000000u) ? (u & 0x7FFFFFFFu) : ~u;
    return __uint_as_float(b);
}

// ---------------------------------------------------------------------------
// K0: init all scratch that is later read.
// ---------------------------------------------------------------------------
__global__ void init_kernel(int* __restrict__ nbrs, int* __restrict__ cnt2,
                            int* __restrict__ cluster, int* __restrict__ isrep,
                            int* __restrict__ count, int* __restrict__ mcount,
                            int* __restrict__ bpool, unsigned* __restrict__ xenc,
                            long long total, int n, long long nd, long long nh)
{
    long long stride = (long long)gridDim.x * blockDim.x;
    for (long long idx = (long long)blockIdx.x * blockDim.x + threadIdx.x;
         idx < total; idx += stride) {
        if (idx < nd) nbrs[idx] = -1;
        if (idx < nh) xenc[idx] = 0u;
        if (idx < n) {
            cnt2[idx]   = 0;
            cluster[idx] = (int)idx;
            isrep[idx]  = 0;
            count[idx]  = 0;
            mcount[idx] = 0;
            bpool[idx]  = 0;
        }
    }
}

// ---------------------------------------------------------------------------
// K1: padded adjacency from the symmetrized edge list (stable first half).
// ---------------------------------------------------------------------------
__global__ void build_adj_kernel(const int* __restrict__ ei, const float* __restrict__ ew,
                                 int* __restrict__ nbrs, float* __restrict__ wts,
                                 int* __restrict__ cnt2, int E, int K, int halfE)
{
    int e = blockIdx.x * blockDim.x + threadIdx.x;
    if (e >= E) return;
    int s = ei[e];
    int d = ei[E + e];
    int slot;
    if (e < halfE) slot = e - s * K;                 // e/K == s by construction
    else           slot = K + atomicAdd(&cnt2[s], 1);
    if (slot < DSLOT) {
        nbrs[(size_t)s * DSLOT + slot] = d;
        wts [(size_t)s * DSLOT + slot] = ew[e];
    }
}

// ---------------------------------------------------------------------------
// K2: serial greedy matching, one wave32, matched bitset in LDS.
// ---------------------------------------------------------------------------
__global__ __launch_bounds__(32)
void graclus_kernel(const int* __restrict__ nbrs, const float* __restrict__ wts,
                    int* __restrict__ cluster, int n)
{
    __shared__ unsigned matched[MAXN_BITS];
    const int lane = threadIdx.x;
    for (int i = lane; i < MAXN_BITS; i += 32) matched[i] = 0u;
    __syncthreads();

    const float NEG_INF = -__builtin_huge_valf();
    for (int u = 0; u < n; ++u) {
        unsigned mu = (matched[u >> 5] >> (u & 31)) & 1u;
        if (!mu) {
            float bw = NEG_INF; int bn = -1; int bs = 0x7FFFFFFF;
#pragma unroll
            for (int s = 0; s < 2; ++s) {
                int slot = lane + 32 * s;
                int nb = nbrs[(size_t)u * DSLOT + slot];
                bool ok = (nb >= 0) && (nb != u);
                if (ok) ok = (((matched[nb >> 5] >> (nb & 31)) & 1u) == 0u);
                if (ok) {
                    float w = wts[(size_t)u * DSLOT + slot];
                    if (w > bw || (w == bw && slot < bs)) { bw = w; bn = nb; bs = slot; }
                }
            }
            for (int off = 16; off > 0; off >>= 1) {
                float ow = __shfl_xor(bw, off, 32);
                int   on = __shfl_xor(bn, off, 32);
                int   os = __shfl_xor(bs, off, 32);
                if ((on >= 0) && (bn < 0 || ow > bw || (ow == bw && os < bs))) {
                    bw = ow; bn = on; bs = os;
                }
            }
            if (lane == 0) {
                matched[u >> 5] |= (1u << (u & 31));
                if (bn >= 0) {
                    matched[bn >> 5] |= (1u << (bn & 31));
                    cluster[bn] = u;
                }
            }
        }
        __syncthreads();
    }
}

// ---------------------------------------------------------------------------
// K3: mark cluster representatives.
// ---------------------------------------------------------------------------
__global__ void mark_reps_kernel(const int* __restrict__ cluster, int* __restrict__ isrep, int n)
{
    int i = blockIdx.x * blockDim.x + threadIdx.x;
    if (i < n) isrep[cluster[i]] = 1;
}

// ---------------------------------------------------------------------------
// K4: single-WG inclusive scan -> newid = cumsum(isrep) - 1.
// ---------------------------------------------------------------------------
__global__ __launch_bounds__(1024)
void scan_kernel(const int* __restrict__ isrep, int* __restrict__ newid, int n)
{
    __shared__ int sh[1024];
    __shared__ int carry;
    if (threadIdx.x == 0) carry = 0;
    __syncthreads();
    for (int base = 0; base < n; base += 1024) {
        int i = base + (int)threadIdx.x;
        int v = (i < n) ? isrep[i] : 0;
        sh[threadIdx.x] = v;
        __syncthreads();
        for (int off = 1; off < 1024; off <<= 1) {
            int t = (threadIdx.x >= (unsigned)off) ? sh[threadIdx.x - off] : 0;
            __syncthreads();
            sh[threadIdx.x] += t;
            __syncthreads();
        }
        if (i < n) newid[i] = carry + sh[threadIdx.x] - 1;
        __syncthreads();
        if (threadIdx.x == 1023) carry += sh[1023];
        __syncthreads();
    }
}

// ---------------------------------------------------------------------------
// K5: relabel, counts, member lists, pooled batch (int max).
// ---------------------------------------------------------------------------
__global__ void assign_kernel(const int* __restrict__ cluster, const int* __restrict__ newid,
                              const int* __restrict__ batch,
                              int* __restrict__ c, int* __restrict__ count,
                              int* __restrict__ mcount, int* __restrict__ members,
                              int* __restrict__ bpool, int n)
{
    int i = blockIdx.x * blockDim.x + threadIdx.x;
    if (i >= n) return;
    int ci = newid[cluster[i]];
    c[i] = ci;
    atomicAdd(&count[ci], 1);
    int sl = atomicAdd(&mcount[ci], 1);
    if (sl < 2) members[ci * 2 + sl] = i;
    atomicMax(&bpool[ci], batch[i]);
}

// ---------------------------------------------------------------------------
// K6: segment-max of features via order-independent uint encoding.
// ---------------------------------------------------------------------------
__global__ void pool_max_kernel(const float* __restrict__ h, const int* __restrict__ c,
                                unsigned* __restrict__ xenc, long long total)
{
    long long idx = (long long)blockIdx.x * blockDim.x + threadIdx.x;
    if (idx >= total) return;
    int i = (int)(idx / HID);
    int f = (int)(idx % HID);
    atomicMax(&xenc[(size_t)c[i] * HID + f], enc_f32(h[idx]));
}

// ---------------------------------------------------------------------------
// K7: decode pooled maxima -> f32 (in place) and f16 copy for WMMA.
// ---------------------------------------------------------------------------
__global__ void decode_kernel(unsigned* __restrict__ xenc, half_t* __restrict__ xh,
                              const int* __restrict__ count, long long total)
{
    long long idx = (long long)blockIdx.x * blockDim.x + threadIdx.x;
    if (idx >= total) return;
    int row = (int)(idx / HID);
    float v = (count[row] > 0) ? dec_f32(xenc[idx]) : 0.0f;
    ((float*)xenc)[idx] = v;
    xh[idx] = (half_t)v;
}

// ---------------------------------------------------------------------------
// K8: BcatT[n][k] (384 x 768, f16): [W_rel | W_root] rows concatenated.
// ---------------------------------------------------------------------------
__global__ void prepB_kernel(const float* __restrict__ Wrel, const float* __restrict__ Wroot,
                             half_t* __restrict__ BT)
{
    int idx = blockIdx.x * blockDim.x + threadIdx.x;
    if (idx >= HID * 2 * HID) return;
    int nn = idx / (2 * HID);
    int k  = idx % (2 * HID);
    float v = (k < HID) ? Wrel[(size_t)nn * HID + k]
                        : Wroot[(size_t)nn * HID + (k - HID)];
    BT[idx] = (half_t)v;
}

// ---------------------------------------------------------------------------
// K9: deterministic deduped neighbor-cluster sum (symmetric relation ->
// gather by source). One WG of 128 threads per cluster row.
// ---------------------------------------------------------------------------
__global__ __launch_bounds__(128)
void aggregate_kernel(const int* __restrict__ nbrs, const int* __restrict__ c,
                      const int* __restrict__ count, const int* __restrict__ mcount,
                      const int* __restrict__ members, const float* __restrict__ xbuf,
                      half_t* __restrict__ aggrh)
{
    const int q = blockIdx.x;
    const int t = threadIdx.x;
    __shared__ int ids[128];
    __shared__ int sorted[128];

    if (count[q] == 0) {
        for (int f = t; f < HID; f += 128) aggrh[(size_t)q * HID + f] = (half_t)0.0f;
        return;
    }
    const int half = t >> 6;
    const int sl = t & 63;
    int cid = INVALID_ID;
    if (half < mcount[q]) {
        int m = members[q * 2 + half];
        int nb = nbrs[(size_t)m * DSLOT + sl];
        if (nb >= 0) {
            int cc = c[nb];
            if (cc != q) cid = cc;
        }
    }
    ids[t] = cid;
    sorted[t] = INVALID_ID;
    __syncthreads();

    int myid = cid;
    if (myid != INVALID_ID) {
        for (int s = 0; s < t; ++s) {
            if (ids[s] == myid) { myid = INVALID_ID; break; }
        }
    }
    __syncthreads();
    ids[t] = myid;
    __syncthreads();

    if (myid != INVALID_ID) {
        int r = 0;
        for (int s = 0; s < 128; ++s) {
            int o = ids[s];
            if (o != INVALID_ID && o < myid) ++r;
        }
        sorted[r] = myid;
    }
    __syncthreads();

    float a0 = 0.f, a1 = 0.f, a2 = 0.f;   // ascending-id order: deterministic
    for (int j = 0; j < 128; ++j) {
        int p = sorted[j];
        if (p == INVALID_ID) break;
        const float* row = xbuf + (size_t)p * HID;
        a0 += row[t];
        a1 += row[t + 128];
        a2 += row[t + 256];
    }
    aggrh[(size_t)q * HID + t]       = (half_t)a0;
    aggrh[(size_t)q * HID + t + 128] = (half_t)a1;
    aggrh[(size_t)q * HID + t + 256] = (half_t)a2;
}

// ---------------------------------------------------------------------------
// K10: fused WMMA GEMM: out = relu([aggr|x] @ [Wrel|Wroot]^T + b_rel),
// masked by count>0. Block tile 128x64, 4 waves x (32x64), K=768, 24 steps,
// fully unrolled. Tiles staged with GLOBAL_LOAD_ASYNC_TO_LDS_B128 (ASYNCcnt),
// double-buffered; fragments via ds_load_b128 per ISA 7.12.2 layouts.
// ---------------------------------------------------------------------------
__global__ __launch_bounds__(128)
void gemm_relu_kernel(const half_t* __restrict__ Aa,   // aggrh [n x 384]
                      const half_t* __restrict__ Ax,   // xh    [n x 384]
                      const half_t* __restrict__ BT,   // BcatT [384 x 768]
                      const float* __restrict__ bias,  // b_rel [384]
                      const int* __restrict__ count,
                      float* __restrict__ out, int n)
{
    __shared__ half_t As[2][128][LDSPAD];
    __shared__ half_t Bs[2][64][LDSPAD];
    const int tid  = threadIdx.x;
    const int wave = tid >> 5;
    const int lane = tid & 31;
    const int rowBase = blockIdx.x * 128;
    const int colBase = blockIdx.y * 64;

    // staging coords: each thread moves 64B of A and 32B of B per k-step
    const int rT  = tid >> 1;            // 0..63
    const int cbT = (tid & 1) << 4;      // 0 or 16 (halfs)
    int growA0 = rowBase + rT;
    int growA1 = rowBase + 64 + rT;
    if (growA0 >= n) growA0 = (n > 0) ? (n - 1) : 0;   // clamped rows never stored
    if (growA1 >= n) growA1 = (n > 0) ? (n - 1) : 0;

    v8f acc[2][4];
#pragma unroll
    for (int m = 0; m < 2; ++m)
#pragma unroll
        for (int j = 0; j < 4; ++j)
#pragma unroll
            for (int e = 0; e < 8; ++e) acc[m][j][e] = 0.0f;

    auto stage = [&](int b, int kt) {
        const half_t* Asrc = (kt < 12) ? Aa : Ax;
        const int kOff = (kt < 12) ? kt * 32 : (kt - 12) * 32;
        const half_t* ga0 = Asrc + (size_t)growA0 * HID + kOff + cbT;
        const half_t* ga1 = Asrc + (size_t)growA1 * HID + kOff + cbT;
        const half_t* gb  = BT + (size_t)(colBase + rT) * (2 * HID) + kt * 32 + cbT;
        unsigned la0 = (unsigned)(size_t)&As[b][rT][cbT];
        unsigned la1 = (unsigned)(size_t)&As[b][64 + rT][cbT];
        unsigned lb  = (unsigned)(size_t)&Bs[b][rT][cbT];
        // both LDS and global sides add the same INST_OFFSET (ISA async pseudocode)
        asm volatile(
            "global_load_async_to_lds_b128 %0, %1, off\n\t"
            "global_load_async_to_lds_b128 %0, %1, off offset:16\n\t"
            "global_load_async_to_lds_b128 %2, %3, off\n\t"
            "global_load_async_to_lds_b128 %2, %3, off offset:16\n\t"
            "global_load_async_to_lds_b128 %4, %5, off\n\t"
            "global_load_async_to_lds_b128 %4, %5, off offset:16"
            :: "v"(la0), "v"(ga0), "v"(la1), "v"(ga1), "v"(lb), "v"(gb)
            : "memory");
    };

    stage(0, 0);
#pragma unroll
    for (int kt = 0; kt < 24; ++kt) {
        const int b = kt & 1;
        if (kt + 1 < 24) {
            stage(b ^ 1, kt + 1);
            asm volatile("s_wait_asynccnt 0x6" ::: "memory");
        } else {
            asm volatile("s_wait_asynccnt 0x0" ::: "memory");
        }
        __syncthreads();

        const int r  = lane & 15;
        const int hf = lane >> 4;
        union { v16h v; v8h h[2]; } af[2];   // A: lane row r, K = 8*hf / 16+8*hf
#pragma unroll
        for (int m = 0; m < 2; ++m) {
            const int row = wave * 32 + m * 16 + r;
            af[m].h[0] = *(const v8h*)&As[b][row][8 * hf];
            af[m].h[1] = *(const v8h*)&As[b][row][16 + 8 * hf];
        }
#pragma unroll
        for (int j = 0; j < 4; ++j) {
            union { v16h v; v8h h[2]; } bf;  // B: lane col n, K = 8*hf / 16+8*hf
            bf.h[0] = *(const v8h*)&Bs[b][j * 16 + r][8 * hf];
            bf.h[1] = *(const v8h*)&Bs[b][j * 16 + r][16 + 8 * hf];
#pragma unroll
            for (int m = 0; m < 2; ++m)
                acc[m][j] = __builtin_amdgcn_wmma_f32_16x16x32_f16(
                    false, af[m].v, false, bf.v, (short)0, acc[m][j], false, false);
        }
        __syncthreads();
    }

    // epilogue: bias + relu + nonempty mask; C/D: lane n=L%16, m=v+8*(L/16)
    const int nloc = lane & 15;
    const int hf = lane >> 4;
#pragma unroll
    for (int m = 0; m < 2; ++m) {
#pragma unroll
        for (int j = 0; j < 4; ++j) {
            const int gcol = colBase + j * 16 + nloc;
            const float bv = bias[gcol];
#pragma unroll
            for (int v = 0; v < 8; ++v) {
                int grow = rowBase + wave * 32 + m * 16 + v + 8 * hf;
                if (grow < n) {
                    float val = acc[m][j][v] + bv;
                    val = (val > 0.f) ? val : 0.f;
                    if (count[grow] == 0) val = 0.f;
                    out[(size_t)grow * HID + gcol] = val;
                }
            }
        }
    }
}

// ---------------------------------------------------------------------------
// K11: pooled batch vector -> tail of d_out (as float).
// ---------------------------------------------------------------------------
__global__ void bpool_out_kernel(const int* __restrict__ bpool, const int* __restrict__ count,
                                 float* __restrict__ outB, int n)
{
    int i = blockIdx.x * blockDim.x + threadIdx.x;
    if (i < n) outB[i] = (count[i] > 0) ? (float)bpool[i] : 0.0f;
}

// ---------------------------------------------------------------------------
// Host launcher.
// ---------------------------------------------------------------------------
static inline size_t align_up(size_t x) { return (x + 255) & ~(size_t)255; }

extern "C" void kernel_launch(void* const* d_in, const int* in_sizes, int n_in,
                              void* d_out, int out_size, void* d_ws, size_t ws_size,
                              hipStream_t stream)
{
    const float* h        = (const float*)d_in[0];
    const int*   ei       = (const int*)d_in[1];
    const float* ew       = (const float*)d_in[2];
    const int*   batch    = (const int*)d_in[3];
    const float* W_rel    = (const float*)d_in[4];
    const float* b_rel    = (const float*)d_in[5];
    const float* W_root   = (const float*)d_in[6];

    const int N = in_sizes[3];
    const int E = in_sizes[2];
    const int K = (E / 2) / N;
    const long long NH = (long long)N * HID;
    const long long ND = (long long)N * DSLOT;

    char* ws = (char*)d_ws;
    size_t cur = 0;
    auto take = [&](size_t bytes) { size_t o = cur; cur = align_up(cur + bytes); return o; };
    int*      nbrs    = (int*)     (ws + take((size_t)ND * 4));
    float*    wts     = (float*)   (ws + take((size_t)ND * 4));
    int*      cnt2    = (int*)     (ws + take((size_t)N * 4));
    int*      cluster = (int*)     (ws + take((size_t)N * 4));
    int*      isrep   = (int*)     (ws + take((size_t)N * 4));
    int*      newid   = (int*)     (ws + take((size_t)N * 4));
    int*      c       = (int*)     (ws + take((size_t)N * 4));
    int*      count   = (int*)     (ws + take((size_t)N * 4));
    int*      mcount  = (int*)     (ws + take((size_t)N * 4));
    int*      members = (int*)     (ws + take((size_t)N * 8));
    int*      bpool   = (int*)     (ws + take((size_t)N * 4));
    unsigned* xenc    = (unsigned*)(ws + take((size_t)NH * 4));   // aliased as f32 x
    half_t*   xh      = (half_t*)  (ws + take((size_t)NH * 2));
    half_t*   aggrh   = (half_t*)  (ws + take((size_t)NH * 2));
    half_t*   BT      = (half_t*)  (ws + take((size_t)HID * 2 * HID * 2));
    (void)ws_size; (void)n_in; (void)out_size;

    float* out  = (float*)d_out;
    float* outB = out + NH;

    const int TB = 256;
    {
        long long total = NH;
        int blocks = (int)((total + TB - 1) / TB);
        hipLaunchKernelGGL(init_kernel, dim3(blocks), dim3(TB), 0, stream,
                           nbrs, cnt2, cluster, isrep, count, mcount, bpool, xenc,
                           total, N, ND, NH);
    }
    hipLaunchKernelGGL(build_adj_kernel, dim3((E + TB - 1) / TB), dim3(TB), 0, stream,
                       ei, ew, nbrs, wts, cnt2, E, K, E / 2);
    hipLaunchKernelGGL(graclus_kernel, dim3(1), dim3(32), 0, stream, nbrs, wts, cluster, N);
    hipLaunchKernelGGL(mark_reps_kernel, dim3((N + TB - 1) / TB), dim3(TB), 0, stream,
                       cluster, isrep, N);
    hipLaunchKernelGGL(scan_kernel, dim3(1), dim3(1024), 0, stream, isrep, newid, N);
    hipLaunchKernelGGL(assign_kernel, dim3((N + TB - 1) / TB), dim3(TB), 0, stream,
                       cluster, newid, batch, c, count, mcount, members, bpool, N);
    hipLaunchKernelGGL(pool_max_kernel, dim3((int)((NH + TB - 1) / TB)), dim3(TB), 0, stream,
                       h, c, xenc, NH);
    hipLaunchKernelGGL(decode_kernel, dim3((int)((NH + TB - 1) / TB)), dim3(TB), 0, stream,
                       xenc, xh, count, NH);
    hipLaunchKernelGGL(prepB_kernel, dim3((HID * 2 * HID + TB - 1) / TB), dim3(TB), 0, stream,
                       W_rel, W_root, BT);
    hipLaunchKernelGGL(aggregate_kernel, dim3(N), dim3(128), 0, stream,
                       nbrs, c, count, mcount, members, (const float*)xenc, aggrh);
    hipLaunchKernelGGL(gemm_relu_kernel, dim3((N + 127) / 128, HID / 64), dim3(128), 0, stream,
                       aggrh, xh, BT, b_rel, count, out, N);
    hipLaunchKernelGGL(bpool_out_kernel, dim3((N + TB - 1) / TB), dim3(TB), 0, stream,
                       bpool, count, outB, N);
}
